// GraphAttentionLayer_56934086476034
// MI455X (gfx1250) — matrix operand order repaired and checked by reading, not compiled, not run on hardware
//
#include <hip/hip_runtime.h>
#include <hip/hip_bf16.h>

typedef __attribute__((ext_vector_type(16))) _Float16 v16h;
typedef __attribute__((ext_vector_type(8)))  float    v8f;

#define BATCH 2
#define NN    2048
#define FIN   256
#define HEADS 8
#define FOUT  32
#define CTOT  (HEADS*FOUT)   // 256
#define NEG_SLOPE 0.01f

// ---------------------------------------------------------------------------
// Kernel 1: Wh = h @ W   (rows = B*N = 4096, K = 256, cols = 256)
// One wave per 16x16 output tile. Stores f16 result TRANSPOSED:
//   Wh_t[((b*HEADS + head)*FOUT + f) * NN + n]
// ---------------------------------------------------------------------------
__global__ void gat_gemm1(const float* __restrict__ h,
                          const float* __restrict__ W,
                          _Float16* __restrict__ Wh_t) {
    const int lane   = threadIdx.x & 31;
    const int hi16   = (lane >= 16) ? 1 : 0;
    const int lmod   = lane & 15;
    const int base   = hi16 ? 8 : 0;        // K sub-block for A operand
    const int rt     = blockIdx.x;          // 0..255 row tiles (flat b*N rows)
    const int ct     = blockIdx.y;          // 0..15 col tiles
    const int arow   = rt * 16 + lmod;      // A row this lane feeds
    const int gcol   = ct * 16 + lmod;      // output column this lane feeds (B operand)
    const float* hrow = h + (size_t)arow * FIN;

    v8f acc = {};
    for (int k0 = 0; k0 < FIN; k0 += 32) {
        // A operand: 16-bit A 16x32 layout. lane<16 -> K {0..7,16..23}, lane>=16 -> {8..15,24..31}
        const float4 a0 = *(const float4*)(hrow + k0 + base);
        const float4 a1 = *(const float4*)(hrow + k0 + base + 4);
        const float4 a2 = *(const float4*)(hrow + k0 + base + 16);
        const float4 a3 = *(const float4*)(hrow + k0 + base + 20);
        float xs[16] = {a0.x,a0.y,a0.z,a0.w, a1.x,a1.y,a1.z,a1.w,
                        a2.x,a2.y,a2.z,a2.w, a3.x,a3.y,a3.z,a3.w};
        v16h av, bv;
        #pragma unroll
        for (int e = 0; e < 16; ++e) av[e] = (_Float16)xs[e];
        // B operand: lane<16 -> col gcol, K {0..7,16..23} ; lane>=16 -> K {8..15,24..31}
        const int kb = k0 + base;
        #pragma unroll
        for (int e = 0; e < 16; ++e) bv[e] = (_Float16)W[(size_t)(kb + (e < 8 ? e : e + 8)) * CTOT + gcol];
        acc = __builtin_amdgcn_wmma_f32_16x16x32_f16(false, av, false, bv,
                                                     (short)0, acc, false, false);
    }

    // D layout: VGPR r holds (M = r + 8*hi16, Ncol = lane%16)
    const int head = gcol >> 5;
    const int f    = gcol & 31;
    #pragma unroll
    for (int r = 0; r < 8; ++r) {
        const int M    = r + (hi16 ? 8 : 0);
        const int row  = rt * 16 + M;           // flat row in [0, B*N)
        const int b    = row >> 11;             // /2048
        const int n    = row & (NN - 1);
        Wh_t[((size_t)((b * HEADS + head) * FOUT + f) << 11) + n] = (_Float16)acc[r];
    }
}

// ---------------------------------------------------------------------------
// Kernel 2: e1[bh][n] = sum_f Wh[b,n,h,f]*a1[f], e2 likewise; mx[bh] = max_n e2
// One block per (b,head) pair.
// ---------------------------------------------------------------------------
__global__ void gat_scores(const _Float16* __restrict__ Wh_t,
                           const float* __restrict__ avec,
                           float* __restrict__ e1,
                           float* __restrict__ e2,
                           float* __restrict__ mx) {
    const int bh = blockIdx.x;       // 0..15
    const int t  = threadIdx.x;      // 0..255
    __shared__ float a1s[FOUT];
    __shared__ float a2s[FOUT];
    __shared__ float red[256];
    if (t < FOUT) { a1s[t] = avec[t]; a2s[t] = avec[FOUT + t]; }
    __syncthreads();

    float lmax = -3.4e38f;
    for (int n = t; n < NN; n += 256) {
        float s1 = 0.f, s2 = 0.f;
        #pragma unroll
        for (int f = 0; f < FOUT; ++f) {
            const float v = (float)Wh_t[((size_t)(bh * FOUT + f) << 11) + n];
            s1 += v * a1s[f];
            s2 += v * a2s[f];
        }
        e1[(bh << 11) + n] = s1;
        e2[(bh << 11) + n] = s2;
        lmax = fmaxf(lmax, s2);
    }
    red[t] = lmax;
    __syncthreads();
    for (int s = 128; s > 0; s >>= 1) {
        if (t < s) red[t] = fmaxf(red[t], red[t + s]);
        __syncthreads();
    }
    if (t == 0) mx[bh] = red[0];
}

// ---------------------------------------------------------------------------
// Kernel 3: fused softmax(leakyrelu(e1_i + e2_j)) @ Wh  -> out (f32)
// One wave per (b, head, 16-row i-tile). K loop over j in chunks of 32.
// max_j lrelu(e1_i+e2_j) == lrelu(e1_i + max_j e2_j)  (monotonicity) -> single pass.
// ---------------------------------------------------------------------------
__device__ __forceinline__ float lrelu_exp(float e2v, float e1v, float mi) {
    float x = e1v + e2v;
    x = fmaxf(x, 0.f) + NEG_SLOPE * fminf(x, 0.f);
    return __expf(x - mi);
}

__global__ void gat_attn(const _Float16* __restrict__ Wh_t,
                         const float* __restrict__ e1,
                         const float* __restrict__ e2,
                         const float* __restrict__ mx,
                         float* __restrict__ out) {
    const int lane  = threadIdx.x & 31;
    const int wave  = threadIdx.x >> 5;
    const int flat  = blockIdx.x * 8 + wave;   // 0..2047
    const int bh    = flat >> 7;               // 0..15
    const int itile = flat & 127;              // 0..127
    const int hi16  = (lane >= 16) ? 1 : 0;
    const int lmod  = lane & 15;
    const int base  = hi16 ? 16 : 0;

    const float e1v = e1[(bh << 11) + itile * 16 + lmod];  // row i = itile*16 + lmod
    const float Mx  = mx[bh];
    float tmi = e1v + Mx;
    const float mi = fmaxf(tmi, 0.f) + NEG_SLOPE * fminf(tmi, 0.f);

    // B operand pointers: lane column f = c*16 + lmod, K rows contiguous in n (transposed V)
    const _Float16* vp0 = Wh_t + ((size_t)(bh * FOUT + lmod)      << 11) + (hi16 ? 16 : 0);
    const _Float16* vp1 = Wh_t + ((size_t)(bh * FOUT + 16 + lmod) << 11) + (hi16 ? 16 : 0);
    const float* e2p = e2 + (bh << 11);

    v8f acc0 = {}, acc1 = {};
    float z = 0.f;
    for (int j0 = 0; j0 < NN; j0 += 32) {
        // e2 values for this lane's 16 A-operand K slots
        const float4 q0 = *(const float4*)(e2p + j0 + base);
        const float4 q1 = *(const float4*)(e2p + j0 + base + 4);
        const float4 q2 = *(const float4*)(e2p + j0 + base + 8);
        const float4 q3 = *(const float4*)(e2p + j0 + base + 12);
        float xs[16] = {q0.x,q0.y,q0.z,q0.w, q1.x,q1.y,q1.z,q1.w,
                        q2.x,q2.y,q2.z,q2.w, q3.x,q3.y,q3.z,q3.w};
        v16h av;
        #pragma unroll
        for (int e = 0; e < 16; ++e) {
            const float p = lrelu_exp(xs[e], e1v, mi);
            z += p;
            av[e] = (_Float16)p;
        }
        const v16h b0 = *(const v16h*)(vp0 + j0);
        const v16h b1 = *(const v16h*)(vp1 + j0);
        acc0 = __builtin_amdgcn_wmma_f32_16x16x32_f16(false, av, false, b0,
                                                      (short)0, acc0, false, false);
        acc1 = __builtin_amdgcn_wmma_f32_16x16x32_f16(false, av, false, b1,
                                                      (short)0, acc1, false, false);
    }

    // Row sums: lane L and L^16 worked on the same row -> combine, then all lanes
    // can fetch any row's Z via shuffle from lane M (M in 0..15).
    z += __shfl_xor(z, 16);

    const int b    = bh >> 3;
    const int head = bh & 7;
    const int n0   = itile * 16;
    #pragma unroll
    for (int r = 0; r < 8; ++r) {
        const int   Mrow = r + (hi16 ? 8 : 0);
        const float zr   = __shfl(z, Mrow);
        const float inv  = 1.f / zr;
        const int   n    = n0 + Mrow;
        float* orow = out + ((size_t)(b * NN + n)) * CTOT + head * FOUT;
        orow[lmod]      = acc0[r] * inv;
        orow[16 + lmod] = acc1[r] * inv;
    }
}

// ---------------------------------------------------------------------------
extern "C" void kernel_launch(void* const* d_in, const int* in_sizes, int n_in,
                              void* d_out, int out_size, void* d_ws, size_t ws_size,
                              hipStream_t stream) {
    const float* h   = (const float*)d_in[0];   // (B, N, FIN)
    const float* adj = (const float*)d_in[1];   // unused by the reference math
    const float* W   = (const float*)d_in[2];   // (FIN, H*F)
    const float* a   = (const float*)d_in[3];   // (2*F, 1)
    (void)adj; (void)in_sizes; (void)n_in; (void)out_size; (void)ws_size;
    float* out = (float*)d_out;                 // (B, N, H*F)

    // Workspace carve-up (~2.26 MB)
    _Float16* Wh_t = (_Float16*)d_ws;                                   // B*H*F*N f16 = 2 MB
    float* e1 = (float*)((char*)d_ws + (size_t)BATCH * CTOT * NN * sizeof(_Float16));
    float* e2 = e1 + BATCH * HEADS * NN;                                // 128 KB each
    float* mx = e2 + BATCH * HEADS * NN;                                // 16 floats

    gat_gemm1<<<dim3((BATCH * NN) / 16, CTOT / 16), 32, 0, stream>>>(h, W, Wh_t);
    gat_scores<<<BATCH * HEADS, 256, 0, stream>>>(Wh_t, a, e1, e2, mx);
    gat_attn<<<(BATCH * HEADS * (NN / 16)) / 8, 256, 0, stream>>>(Wh_t, e1, e2, mx, out);
}
